// TransLayer_77300821394013
// MI455X (gfx1250) — matrix operand order repaired and checked by reading, not compiled, not run on hardware
//
#include <hip/hip_runtime.h>
#include <hip/hip_bf16.h>

// ---------------------------------------------------------------------------
// Nystromformer layer for MI455X (gfx1250, wave32, WMMA).
//   bf16 WMMA (v_wmma_f32_16x16x32_bf16) for all large GEMMs (f32 accum),
//     with cooperative double-buffered LDS staging (async-to-LDS if the
//     toolchain exposes GLOBAL_LOAD_ASYNC_TO_LDS_B128).
//   f32  WMMA (v_wmma_f32_16x16x4_f32) for the Newton-Schulz pinv chain.
// Workspace requirement: ~930 MB (laid out below with aliasing).
// ---------------------------------------------------------------------------

typedef __attribute__((ext_vector_type(16))) __bf16 v16bf;
typedef __attribute__((ext_vector_type(8)))  __bf16 v8bf;
typedef __attribute__((ext_vector_type(8)))  float  v8f;
typedef __attribute__((ext_vector_type(2)))  float  v2f;
typedef __attribute__((ext_vector_type(4)))  int    v4i;

constexpr int  B_   = 8;
constexpr int  N_   = 4096;
constexpr int  D_   = 512;
constexpr int  H_   = 8;
constexpr int  DH_  = 64;
constexpr int  M_   = 256;
constexpr int  L_   = 16;     // tokens per landmark
constexpr int  KER_ = 33;
constexpr int  BH_  = B_ * H_;          // 64
constexpr long BN_  = (long)B_ * N_;    // 32768

#if defined(__AMDGCN__) && __has_builtin(__builtin_amdgcn_global_load_async_to_lds_b128)
#define USE_ASYNC_LDS 1
typedef __attribute__((address_space(1))) v4i* gas_v4i;   // global int4*
typedef __attribute__((address_space(3))) v4i* las_v4i;   // LDS int4*
__device__ __forceinline__ void async_g2l_b128(const void* g, void* l) {
  // AS1 ptr == generic 64-bit address; AS3 ptr == low 32 bits of generic
  // shared pointer (ISA 10.2).
  __builtin_amdgcn_global_load_async_to_lds_b128(
      (gas_v4i)(unsigned long long)g,
      (las_v4i)(unsigned)(unsigned long long)l, 0, 0);
}
__device__ __forceinline__ void wait_async_lds() {
#if __has_builtin(__builtin_amdgcn_s_wait_asynccnt)
  __builtin_amdgcn_s_wait_asynccnt(0);
#else
  asm volatile("s_wait_asynccnt 0" ::: "memory");
#endif
}
#else
#define USE_ASYNC_LDS 0
#endif

// ------------------------------ LayerNorm ----------------------------------
__global__ void k_layernorm(const float* __restrict__ x,
                            const float* __restrict__ gamma,
                            const float* __restrict__ beta,
                            __bf16* __restrict__ xn) {
  __shared__ float sm[256];
  const int  t   = threadIdx.x;
  const long row = blockIdx.x;
  const float* xr = x + row * D_;
  float a = xr[t], b = xr[t + 256];
  sm[t] = a + b; __syncthreads();
  for (int off = 128; off > 0; off >>= 1) { if (t < off) sm[t] += sm[t + off]; __syncthreads(); }
  float mu = sm[0] * (1.0f / D_); __syncthreads();
  float da = a - mu, db = b - mu;
  sm[t] = da * da + db * db; __syncthreads();
  for (int off = 128; off > 0; off >>= 1) { if (t < off) sm[t] += sm[t + off]; __syncthreads(); }
  float rstd = rsqrtf(sm[0] * (1.0f / D_) + 1e-5f);
  __bf16* o = xn + row * D_;
  o[t]       = (__bf16)(da * rstd * gamma[t]       + beta[t]);
  o[t + 256] = (__bf16)(db * rstd * gamma[t + 256] + beta[t + 256]);
}

// ------------------------- f32 -> bf16 convert -----------------------------
__global__ void k_cvt_bf16(const float* __restrict__ s, __bf16* __restrict__ d, long n) {
  long i = blockIdx.x * 256L + threadIdx.x;
  if (i < n) d[i] = (__bf16)s[i];
}

// --------------- bf16 NT GEMM, LDS-staged (WMMA 16x16x32) ------------------
// C[z][M][N] = A[z][M][K] * B[z][N][K]^T  (+bias[N]) (+resid) (or += C)
// Block = 256 thr (8 waves), block tile 128(M) x 64(N), K-step 32.
// Waves arranged 4(M) x 2(N); each wave computes a 32x32 sub-tile via
// 2x2 WMMA fragments, with A/B slices double-buffered in LDS.
// grid = (N/64, M/128, batch). Requires M%128==0, N%64==0, K%32==0, K>=32.
#define LDPAD 40   // LDS row stride in elements (80 B = 20 banks; half-wave
                   // fragment reads land on distinct banks, spacing 4)

template <bool C_BF16, bool ACCUM, bool BIAS, bool RESID>
__global__ void k_gemm_bf16_nt(const __bf16* __restrict__ A,
                               const __bf16* __restrict__ B,
                               void* __restrict__ Cv,
                               const float* __restrict__ bias,
                               const float* __restrict__ resid,
                               int Mi, int Ni, int Ki,
                               long sA, long sB, long sC) {
  const int tid  = threadIdx.x;
  const int lane = tid & 31;
  const int wave = tid >> 5;
  const int wm   = wave >> 1;           // 0..3
  const int wn   = wave & 1;            // 0..1
  const int n0   = blockIdx.x << 6;     // block N origin
  const int m0   = blockIdx.y << 7;     // block M origin
  const long zb  = blockIdx.z;
  float accv[2][2][8];
#if defined(__AMDGCN__) && __has_builtin(__builtin_amdgcn_wmma_f32_16x16x32_bf16)
  __shared__ __bf16 smA[2][128 * LDPAD];
  __shared__ __bf16 smB[2][64 * LDPAD];
  const __bf16* Ag = A + zb * sA + (long)m0 * Ki;
  const __bf16* Bg = B + zb * sB + (long)n0 * Ki;
  // staging map: 3 x 16B chunks per thread; chunk c -> row=c>>2, col=(c&3)*8
  int srow[3]; long goff[3]; int loff[3]; bool isA[3];
#pragma unroll
  for (int t = 0; t < 3; ++t) {
    int c  = tid + t * 256;
    int r  = c >> 2, ch = c & 3;
    isA[t] = (r < 128);
    srow[t] = isA[t] ? r : (r - 128);
    goff[t] = (long)srow[t] * Ki + ch * 8;
    loff[t] = srow[t] * LDPAD + ch * 8;
  }
  const int ca = (lane >> 4) << 3;   // A frag chunk base (0 / 8)
  const int cb = (lane >> 4) << 3;   // B frag chunk base (0 / 8)
  const int ar = (lane & 15);
  v8f acc[2][2];
#pragma unroll
  for (int mt = 0; mt < 2; ++mt)
#pragma unroll
    for (int nt = 0; nt < 2; ++nt) acc[mt][nt] = (v8f){};

#if USE_ASYNC_LDS
  // prologue: async-stage k=0 into buffer 0
#pragma unroll
  for (int t = 0; t < 3; ++t) {
    const __bf16* g = (isA[t] ? Ag : Bg) + goff[t];
    __bf16* l = (isA[t] ? &smA[0][0] : &smB[0][0]) + loff[t];
    async_g2l_b128(g, l);
  }
  wait_async_lds();
  __syncthreads();
#else
  {
    uint4 rg[3];
#pragma unroll
    for (int t = 0; t < 3; ++t)
      rg[t] = *(const uint4*)((isA[t] ? Ag : Bg) + goff[t]);
#pragma unroll
    for (int t = 0; t < 3; ++t)
      *(uint4*)((isA[t] ? &smA[0][0] : &smB[0][0]) + loff[t]) = rg[t];
  }
  __syncthreads();
#endif

  int cur = 0;
  for (int k = 0; k < Ki; k += 32) {
    const bool more = (k + 32 < Ki);
#if USE_ASYNC_LDS
    if (more) {   // async DMA into the other buffer; overlaps with WMMA below
#pragma unroll
      for (int t = 0; t < 3; ++t) {
        const __bf16* g = (isA[t] ? Ag : Bg) + goff[t] + (k + 32);
        __bf16* l = (isA[t] ? &smA[cur ^ 1][0] : &smB[cur ^ 1][0]) + loff[t];
        async_g2l_b128(g, l);
      }
    }
#else
    uint4 rg[3];
    if (more) {   // issue global loads now; store after compute
#pragma unroll
      for (int t = 0; t < 3; ++t)
        rg[t] = *(const uint4*)((isA[t] ? Ag : Bg) + goff[t] + (k + 32));
    }
#endif
    // ---- compute from buffer `cur` ----
    const __bf16* As = &smA[cur][0];
    const __bf16* Bs = &smB[cur][0];
    v16bf af[2], bf[2];
#pragma unroll
    for (int mt = 0; mt < 2; ++mt) {
      const __bf16* ap = As + (wm * 32 + mt * 16 + ar) * LDPAD + ca;
      v8bf lo = *(const v8bf*)(ap);
      v8bf hi = *(const v8bf*)(ap + 16);
      af[mt] = __builtin_shufflevector(lo, hi, 0,1,2,3,4,5,6,7,8,9,10,11,12,13,14,15);
    }
#pragma unroll
    for (int nt = 0; nt < 2; ++nt) {
      const __bf16* bp = Bs + (wn * 32 + nt * 16 + ar) * LDPAD + cb;
      v8bf lo = *(const v8bf*)(bp);
      v8bf hi = *(const v8bf*)(bp + 16);
      bf[nt] = __builtin_shufflevector(lo, hi, 0,1,2,3,4,5,6,7,8,9,10,11,12,13,14,15);
    }
#pragma unroll
    for (int mt = 0; mt < 2; ++mt)
#pragma unroll
      for (int nt = 0; nt < 2; ++nt)
        acc[mt][nt] = __builtin_amdgcn_wmma_f32_16x16x32_bf16(
            false, af[mt], false, bf[nt], (short)0, acc[mt][nt], false, false);
    // ---- publish next buffer ----
    if (more) {
#if USE_ASYNC_LDS
      wait_async_lds();
#else
#pragma unroll
      for (int t = 0; t < 3; ++t)
        *(uint4*)((isA[t] ? &smA[cur ^ 1][0] : &smB[cur ^ 1][0]) + loff[t]) = rg[t];
#endif
    }
    __syncthreads();
    cur ^= 1;
  }
#pragma unroll
  for (int mt = 0; mt < 2; ++mt)
#pragma unroll
    for (int nt = 0; nt < 2; ++nt)
#pragma unroll
      for (int r = 0; r < 8; ++r) accv[mt][nt][r] = acc[mt][nt][r];
#else
  // scalar fallback (numerically identical; host-parse / no-builtin path)
  const __bf16* Ab = A + zb * sA;
  const __bf16* Bp = B + zb * sB;
#pragma unroll
  for (int mt = 0; mt < 2; ++mt)
#pragma unroll
    for (int nt = 0; nt < 2; ++nt)
#pragma unroll
      for (int r = 0; r < 8; ++r) accv[mt][nt][r] = 0.0f;
  for (int k = 0; k < Ki; ++k) {
    for (int mt = 0; mt < 2; ++mt)
      for (int nt = 0; nt < 2; ++nt)
        for (int r = 0; r < 8; ++r) {
          int row = m0 + wm * 32 + mt * 16 + ((lane >> 4) << 3) + r;
          int col = n0 + wn * 32 + nt * 16 + (lane & 15);
          accv[mt][nt][r] = fmaf((float)Ab[(long)row * Ki + k],
                                 (float)Bp[(long)col * Ki + k], accv[mt][nt][r]);
        }
  }
#endif
  const long cbase = zb * sC;
#pragma unroll
  for (int mt = 0; mt < 2; ++mt)
#pragma unroll
    for (int nt = 0; nt < 2; ++nt) {
      const int col = n0 + wn * 32 + nt * 16 + (lane & 15);
      const int rb  = m0 + wm * 32 + mt * 16 + ((lane >> 4) << 3);
#pragma unroll
      for (int r = 0; r < 8; ++r) {
        long idx = cbase + (long)(rb + r) * Ni + col;
        float val = accv[mt][nt][r];
        if (RESID) val += resid[idx];
        if (BIAS)  val += bias[col];
        if (ACCUM)       ((float*)Cv)[idx] += val;
        else if (C_BF16) ((__bf16*)Cv)[idx] = (__bf16)val;
        else             ((float*)Cv)[idx]  = val;
      }
    }
}

// ------------- f32 NT GEMM 256x256, K=64 (sim2, WMMA 16x16x4) --------------
__global__ void k_gemm_f32_nt_64(const float* __restrict__ A,
                                 const float* __restrict__ B,
                                 float* __restrict__ C) {
  const int lane = threadIdx.x & 31;
  const int wave = threadIdx.x >> 5;
  const int n0 = ((blockIdx.x << 2) + wave) << 4;
  const int m0 = blockIdx.y << 4;
  const long bA = (long)blockIdx.z * M_ * DH_;
  const long bC = (long)blockIdx.z * M_ * M_;
  const int colj = n0 + (lane & 15);
  const int rb   = m0 + ((lane >> 4) << 3);
  float accv[8];
#if defined(__AMDGCN__) && __has_builtin(__builtin_amdgcn_wmma_f32_16x16x4_f32)
  const float* Ar = A + bA + (long)(m0 + (lane & 15)) * DH_;
  const float* Br = B + bA + (long)(n0 + (lane & 15)) * DH_;
  const int koff = (lane >> 4) << 1;  // 0 or 2
  v8f acc = {};
  for (int k = 0; k < DH_; k += 4) {
    v2f av = *(const v2f*)(Ar + k + koff);
    v2f bv = *(const v2f*)(Br + k + koff);
    acc = __builtin_amdgcn_wmma_f32_16x16x4_f32(false, av, false, bv,
                                                (short)0, acc, false, false);
  }
#pragma unroll
  for (int r = 0; r < 8; ++r) accv[r] = acc[r];
#else
  const float* Bp = B + bA + (long)colj * DH_;
#pragma unroll
  for (int r = 0; r < 8; ++r) accv[r] = 0.0f;
  for (int k = 0; k < DH_; ++k) {
    float bvv = Bp[k];
#pragma unroll
    for (int r = 0; r < 8; ++r)
      accv[r] = fmaf(A[bA + (long)(rb + r) * DH_ + k], bvv, accv[r]);
  }
#endif
#pragma unroll
  for (int r = 0; r < 8; ++r) C[bC + (long)(rb + r) * M_ + colj] = accv[r];
}

// ------- f32 NN GEMM 256^3 with fused (bc1*I + bc0*B), C = alpha*A*Beff ----
__global__ void k_gemm_f32_nn_ns(const float* __restrict__ A,
                                 const float* __restrict__ Bm,
                                 float* __restrict__ C,
                                 float alpha, float bc0, float bc1) {
  const int lane = threadIdx.x & 31;
  const int wave = threadIdx.x >> 5;
  const int n0 = ((blockIdx.x << 2) + wave) << 4;
  const int m0 = blockIdx.y << 4;
  const long base = (long)blockIdx.z * M_ * M_;
  const int colj = n0 + (lane & 15);
  const int rb   = m0 + ((lane >> 4) << 3);
  float accv[8];
#if defined(__AMDGCN__) && __has_builtin(__builtin_amdgcn_wmma_f32_16x16x4_f32)
  const float* Ar = A + base + (long)(m0 + (lane & 15)) * M_;
  const int koff = (lane >> 4) << 1;
  v8f acc = {};
  for (int k = 0; k < M_; k += 4) {
    const int kk = k + koff;
    v2f av = *(const v2f*)(Ar + kk);
    v2f bv;
    bv.x = bc0 * Bm[base + (long)kk * M_ + colj]       + ((kk == colj)     ? bc1 : 0.0f);
    bv.y = bc0 * Bm[base + (long)(kk + 1) * M_ + colj] + ((kk + 1 == colj) ? bc1 : 0.0f);
    acc = __builtin_amdgcn_wmma_f32_16x16x4_f32(false, av, false, bv,
                                                (short)0, acc, false, false);
  }
#pragma unroll
  for (int r = 0; r < 8; ++r) accv[r] = acc[r];
#else
#pragma unroll
  for (int r = 0; r < 8; ++r) accv[r] = 0.0f;
  for (int k = 0; k < M_; ++k) {
    float be = bc0 * Bm[base + (long)k * M_ + colj] + ((k == colj) ? bc1 : 0.0f);
#pragma unroll
    for (int r = 0; r < 8; ++r)
      accv[r] = fmaf(A[base + (long)(rb + r) * M_ + k], be, accv[r]);
  }
#endif
#pragma unroll
  for (int r = 0; r < 8; ++r) C[base + (long)(rb + r) * M_ + colj] = alpha * accv[r];
}

// ------------------------------- softmax -----------------------------------
template <typename OutT, int EPT>
__global__ void k_softmax(const float* __restrict__ in, OutT* __restrict__ out) {
  __shared__ float sm[256];
  const int  t   = threadIdx.x;
  const long row = blockIdx.x;
  const int  Ld  = EPT * 256;
  const float* ir = in + row * (long)Ld;
  float vals[EPT];
  float mx = -3.0e38f;
#pragma unroll
  for (int e = 0; e < EPT; ++e) { vals[e] = ir[t + e * 256]; mx = fmaxf(mx, vals[e]); }
  sm[t] = mx; __syncthreads();
  for (int off = 128; off > 0; off >>= 1) { if (t < off) sm[t] = fmaxf(sm[t], sm[t + off]); __syncthreads(); }
  mx = sm[0]; __syncthreads();
  float s = 0.0f;
#pragma unroll
  for (int e = 0; e < EPT; ++e) { vals[e] = expf(vals[e] - mx); s += vals[e]; }
  sm[t] = s; __syncthreads();
  for (int off = 128; off > 0; off >>= 1) { if (t < off) sm[t] += sm[t + off]; __syncthreads(); }
  float inv = 1.0f / sm[0];
  OutT* orow = out + row * (long)Ld;
#pragma unroll
  for (int e = 0; e < EPT; ++e) orow[t + e * 256] = (OutT)(vals[e] * inv);
}

// -------------------- split qkv + scale q + transpose v --------------------
__global__ void k_split_qkv(const __bf16* __restrict__ qkv,
                            __bf16* __restrict__ q, __bf16* __restrict__ k,
                            __bf16* __restrict__ v, __bf16* __restrict__ vT) {
  long gid = blockIdx.x * 256L + threadIdx.x;   // [b][h][i][d]
  int d = gid & 63; long t2 = gid >> 6;
  int i = (int)(t2 & 4095); int bh = (int)(t2 >> 12);
  int h = bh & 7; int b = bh >> 3;
  long src = ((long)b * N_ + i) * 1536 + h * DH_ + d;
  q[gid] = (__bf16)((float)qkv[src] * 0.125f);   // * dh^-0.5
  k[gid] = qkv[src + 512];
  __bf16 vv = qkv[src + 1024];
  v[gid] = vv;
  vT[((long)bh * DH_ + d) * N_ + i] = vv;
}

// ------------------------------ landmarks ----------------------------------
__global__ void k_landmarks(const __bf16* __restrict__ q, const __bf16* __restrict__ k,
                            float* __restrict__ qlf, float* __restrict__ klf,
                            __bf16* __restrict__ qlb, __bf16* __restrict__ klb) {
  long gid = blockIdx.x * 256L + threadIdx.x;   // [bh][j][d]
  int d = gid & 63; long t2 = gid >> 6;
  int j = (int)(t2 & 255); int bh = (int)(t2 >> 8);
  long base = ((long)bh * N_ + (long)j * L_) * DH_ + d;
  float sq = 0.0f, sk = 0.0f;
#pragma unroll
  for (int t = 0; t < L_; ++t) {
    sq += (float)q[base + (long)t * DH_];
    sk += (float)k[base + (long)t * DH_];
  }
  sq *= (1.0f / L_); sk *= (1.0f / L_);
  qlf[gid] = sq; klf[gid] = sk;
  qlb[gid] = (__bf16)sq; klb[gid] = (__bf16)sk;
}

// -------------------- Newton-Schulz init helpers ---------------------------
__global__ void k_zero2(unsigned* p) { if (threadIdx.x < 2) p[threadIdx.x] = 0u; }

__global__ void k_ns_absmax(const float* __restrict__ a2, unsigned* __restrict__ gmx) {
  __shared__ float sm[256];
  const int t = threadIdx.x;
  const long base = (long)blockIdx.x * M_ * M_;
  float rs = 0.0f, cs = 0.0f;
  for (int j = 0; j < M_; ++j) {
    rs += fabsf(a2[base + (long)t * M_ + j]);
    cs += fabsf(a2[base + (long)j * M_ + t]);
  }
  sm[t] = rs; __syncthreads();
  for (int off = 128; off > 0; off >>= 1) { if (t < off) sm[t] = fmaxf(sm[t], sm[t + off]); __syncthreads(); }
  if (t == 0) atomicMax(&gmx[0], __float_as_uint(sm[0]));
  __syncthreads();
  sm[t] = cs; __syncthreads();
  for (int off = 128; off > 0; off >>= 1) { if (t < off) sm[t] = fmaxf(sm[t], sm[t + off]); __syncthreads(); }
  if (t == 0) atomicMax(&gmx[1], __float_as_uint(sm[0]));
}

__global__ void k_ns_init_z(const float* __restrict__ a2, float* __restrict__ z,
                            const unsigned* __restrict__ gmx) {
  const float inv = 1.0f / (__uint_as_float(gmx[0]) * __uint_as_float(gmx[1]));
  long gid = blockIdx.x * 256L + threadIdx.x;   // [z][i][j]
  int j = (int)(gid & 255); int i = (int)((gid >> 8) & 255); long zb = gid >> 16;
  z[(zb << 16) + ((long)i << 8) + j] = a2[(zb << 16) + ((long)j << 8) + i] * inv;
}

// -------------------- batched transpose + bf16 convert ---------------------
__global__ void k_cvt_T_bf16(const float* __restrict__ s, __bf16* __restrict__ d,
                             int R, int C) {
  long gid = blockIdx.x * 256L + threadIdx.x;   // gid = ((z*R+r)*C+c)
  int c = (int)(gid % C); long t2 = gid / C;
  int r = (int)(t2 % R);  long z  = t2 / R;
  d[(z * C + c) * (long)R + r] = (__bf16)s[gid];
}

// -------------------- depthwise conv residual over seq ---------------------
__global__ void k_dwconv(const __bf16* __restrict__ v, const float* __restrict__ w,
                         float* __restrict__ o) {
  long gid = blockIdx.x * 256L + threadIdx.x;   // [bh][i][d]
  int d = (int)(gid & 63); int i = (int)((gid >> 6) & 4095);
  int bh = (int)(gid >> 18); int h = bh & 7;
  const __bf16* vb = v + ((long)bh * N_) * DH_ + d;
  float acc = 0.0f;
#pragma unroll
  for (int t = 0; t < KER_; ++t) {
    int ii = i + t - KER_ / 2;
    if (ii >= 0 && ii < N_) acc = fmaf(w[h * KER_ + t], (float)vb[(long)ii * DH_], acc);
  }
  o[gid] = acc;
}

// ----------------------------- merge heads ---------------------------------
__global__ void k_merge_heads(const float* __restrict__ oh, __bf16* __restrict__ mh) {
  long gid = blockIdx.x * 256L + threadIdx.x;   // [b][i][h][d]
  int d = (int)(gid & 63); long t2 = gid >> 6;
  int h = (int)(t2 & 7);   long t3 = t2 >> 3;
  int i = (int)(t3 & 4095); int b = (int)(t3 >> 12);
  mh[gid] = (__bf16)oh[(((long)(b * H_ + h) * N_) + i) * DH_ + d];
}

// ===========================================================================
static inline size_t alignup(size_t x) { return (x + 255) & ~(size_t)255; }

extern "C" void kernel_launch(void* const* d_in, const int* in_sizes, int n_in,
                              void* d_out, int out_size, void* d_ws, size_t ws_size,
                              hipStream_t stream) {
  (void)in_sizes; (void)n_in; (void)out_size; (void)ws_size;
  const float* x     = (const float*)d_in[0];
  const float* gamma = (const float*)d_in[1];
  const float* beta  = (const float*)d_in[2];
  const float* wqkv  = (const float*)d_in[3];
  const float* wout  = (const float*)d_in[4];
  const float* bout  = (const float*)d_in[5];
  const float* wres  = (const float*)d_in[6];
  float* out = (float*)d_out;

  char* ws = (char*)d_ws;
  size_t off = 0;
  auto alloc = [&](size_t bytes) -> char* { char* p = ws + off; off = alignup(off + bytes); return p; };

  const long QKVE = (long)BH_ * N_ * DH_;       // 16.7M elems
  const long MME  = (long)BH_ * M_ * M_;        // 4.2M elems
  const long NME  = (long)BH_ * N_ * M_;        // 67.1M elems

  __bf16* xn     = (__bf16*)alloc(BN_ * D_ * 2);
  __bf16* wqkv_b = (__bf16*)alloc(1536L * 512 * 2);
  __bf16* wout_b = (__bf16*)alloc(512L * 512 * 2);
  __bf16* qkv_b  = (__bf16*)alloc(BN_ * 1536L * 2);
  __bf16* qb     = (__bf16*)alloc(QKVE * 2);
  __bf16* kb     = (__bf16*)alloc(QKVE * 2);
  __bf16* vb     = (__bf16*)alloc(QKVE * 2);
  __bf16* vTb    = (__bf16*)alloc(QKVE * 2);
  float*  qlf    = (float*)alloc((long)BH_ * M_ * DH_ * 4);
  float*  klf    = (float*)alloc((long)BH_ * M_ * DH_ * 4);
  __bf16* qlb    = (__bf16*)alloc((long)BH_ * M_ * DH_ * 2);
  __bf16* klb    = (__bf16*)alloc((long)BH_ * M_ * DH_ * 2);
  float*  a2     = (float*)alloc(MME * 4);
  float*  zA     = (float*)alloc(MME * 4);
  float*  zB     = (float*)alloc(MME * 4);
  float*  xz     = (float*)alloc(MME * 4);
  float*  r1     = (float*)alloc(MME * 4);
  float*  r2     = (float*)alloc(MME * 4);
  __bf16* a2invT = (__bf16*)alloc(MME * 2);
  unsigned* gmx  = (unsigned*)alloc(256);
  float*  logits = (float*)alloc(NME * 4);      // 268 MB, also aliased below
  __bf16* a1b    = (__bf16*)alloc(NME * 2);
  __bf16* a3b    = (__bf16*)alloc(NME * 2);     // reused as t2 (same size)
  // aliases inside the (sequentially dead) logits region:
  float*  t1f  = logits;                                  // [bh][256][64] f32
  __bf16* t1Tb = (__bf16*)((char*)logits + (4L << 20));   // [bh][64][256] bf16
  float*  outh = (float*)((char*)logits + (8L << 20));    // [bh][4096][64] f32
  __bf16* mh   = (__bf16*)((char*)logits + (80L << 20));  // [b*n][512] bf16
  __bf16* t2b  = a3b;

  // 1. LayerNorm -> xn (bf16)
  k_layernorm<<<(int)BN_, 256, 0, stream>>>(x, gamma, beta, xn);
  // 2. weight conversions
  k_cvt_bf16<<<(1536 * 512) / 256, 256, 0, stream>>>(wqkv, wqkv_b, 1536L * 512);
  k_cvt_bf16<<<(512 * 512) / 256, 256, 0, stream>>>(wout, wout_b, 512L * 512);
  // 3. QKV projection: [32768,1536] = xn[32768,512] @ wqkv^T
  k_gemm_bf16_nt<true, false, false, false><<<dim3(1536 / 64, 32768 / 128, 1), 256, 0, stream>>>(
      xn, wqkv_b, qkv_b, nullptr, nullptr, 32768, 1536, 512, 0, 0, 0);
  // 4. split heads, scale q, transpose v
  k_split_qkv<<<(int)(QKVE / 256), 256, 0, stream>>>(qkv_b, qb, kb, vb, vTb);
  // 5. landmark means
  k_landmarks<<<(int)((long)BH_ * M_ * DH_ / 256), 256, 0, stream>>>(qb, kb, qlf, klf, qlb, klb);
  // 6. sim2 = q_l @ k_l^T  (f32 WMMA) ; softmax in-place -> a2
  k_gemm_f32_nt_64<<<dim3(4, 16, BH_), 128, 0, stream>>>(qlf, klf, a2);
  k_softmax<float, 1><<<BH_ * M_, 256, 0, stream>>>(a2, a2);
  // 7. Newton-Schulz init: z0 = a2^T / (max(rowsums)*max(colsums))
  k_zero2<<<1, 32, 0, stream>>>(gmx);
  k_ns_absmax<<<BH_, 256, 0, stream>>>(a2, gmx);
  k_ns_init_z<<<(int)(MME / 256), 256, 0, stream>>>(a2, zA, gmx);
  // 8. 6 Newton-Schulz iterations (4 f32-WMMA GEMMs each)
  float* zi = zA; float* zo = zB;
  for (int it = 0; it < 6; ++it) {
    k_gemm_f32_nn_ns<<<dim3(4, 16, BH_), 128, 0, stream>>>(a2, zi, xz, 1.0f,  1.0f,  0.0f);
    k_gemm_f32_nn_ns<<<dim3(4, 16, BH_), 128, 0, stream>>>(xz, xz, r1, 1.0f, -1.0f,  7.0f);
    k_gemm_f32_nn_ns<<<dim3(4, 16, BH_), 128, 0, stream>>>(xz, r1, r2, 1.0f, -1.0f, 15.0f);
    k_gemm_f32_nn_ns<<<dim3(4, 16, BH_), 128, 0, stream>>>(zi, r2, zo, 0.25f, -1.0f, 13.0f);
    float* tmp = zi; zi = zo; zo = tmp;
  }
  k_cvt_T_bf16<<<(int)(MME / 256), 256, 0, stream>>>(zi, a2invT, M_, M_);
  // 9. sim1 = q @ k_l^T ; softmax -> a1 (bf16)
  k_gemm_bf16_nt<false, false, false, false><<<dim3(4, 4096 / 128, BH_), 256, 0, stream>>>(
      qb, klb, logits, nullptr, nullptr, N_, M_, DH_,
      (long)N_ * DH_, (long)M_ * DH_, (long)N_ * M_);
  k_softmax<__bf16, 1><<<BH_ * N_, 256, 0, stream>>>(logits, a1b);
  // 10. sim3 = q_l @ k^T ; softmax -> a3 (bf16)
  k_gemm_bf16_nt<false, false, false, false><<<dim3(64, 2, BH_), 256, 0, stream>>>(
      qlb, kb, logits, nullptr, nullptr, M_, N_, DH_,
      (long)M_ * DH_, (long)N_ * DH_, (long)M_ * N_);
  k_softmax<__bf16, 16><<<BH_ * M_, 256, 0, stream>>>(logits, a3b);
  // 11. t1 = a3 @ v  -> t1T (bf16)
  k_gemm_bf16_nt<false, false, false, false><<<dim3(1, 2, BH_), 256, 0, stream>>>(
      a3b, vTb, t1f, nullptr, nullptr, M_, DH_, N_,
      (long)M_ * N_, (long)DH_ * N_, (long)M_ * DH_);
  k_cvt_T_bf16<<<(int)((long)BH_ * M_ * DH_ / 256), 256, 0, stream>>>(t1f, t1Tb, M_, DH_);
  // 12. depthwise conv residual -> outh
  k_dwconv<<<(int)(QKVE / 256), 256, 0, stream>>>(vb, wres, outh);
  // 13. t2 = a1 @ a2_inv (bf16, overwrites a3 buffer)
  k_gemm_bf16_nt<true, false, false, false><<<dim3(4, 4096 / 128, BH_), 256, 0, stream>>>(
      a1b, a2invT, t2b, nullptr, nullptr, N_, M_, M_,
      (long)N_ * M_, (long)M_ * M_, (long)N_ * M_);
  // 14. outh += t2 @ t1
  k_gemm_bf16_nt<false, true, false, false><<<dim3(1, 4096 / 128, BH_), 256, 0, stream>>>(
      t2b, t1Tb, outh, nullptr, nullptr, N_, DH_, M_,
      (long)N_ * M_, (long)DH_ * M_, (long)N_ * DH_);
  // 15. merge heads -> mh (bf16)
  k_merge_heads<<<(int)(QKVE / 256), 256, 0, stream>>>(outh, mh);
  // 16. final: out = x + mh @ wout^T + b_out
  k_gemm_bf16_nt<false, false, true, true><<<dim3(512 / 64, 32768 / 128, 1), 256, 0, stream>>>(
      mh, wout_b, out, bout, x, 32768, 512, 512, 0, 0, 0);
}
